// ROITemporalAlignment_76699525972083
// MI455X (gfx1250) — matrix operand order repaired and checked by reading, not compiled, not run on hardware
//
#include <hip/hip_runtime.h>

// ---------------------------------------------------------------------------
// CDNA5 (gfx1250) bf16 WMMA implementation of ROITemporalAlignment forward.
// wave32; V_WMMA_F32_16X16X32_BF16 for all GEMMs, f32 accumulation.
// GLOBAL_LOAD_ASYNC_TO_LDS_B128 (ASYNCcnt) for GEMM B-tile / conv weight-tile
// staging on the bandwidth-critical paths.
// ---------------------------------------------------------------------------

typedef __attribute__((ext_vector_type(16))) __bf16 v16bf;
typedef __attribute__((ext_vector_type(8)))  float  v8f;

union Frag16 { v16bf v; unsigned short u[16]; };

__device__ __forceinline__ unsigned short f2bf(float f) {
  unsigned int u = __float_as_uint(f);
  u += 0x7FFFu + ((u >> 16) & 1u);            // round-to-nearest-even
  return (unsigned short)(u >> 16);
}

// LDS byte offset of a __shared__ object (generic ptr low 32 bits = LDS addr)
__device__ __forceinline__ unsigned lds_off(const void* p) {
  return (unsigned)(size_t)p;
}

// Async copy 16B global -> LDS (tracked by ASYNCcnt, bypasses VGPRs)
__device__ __forceinline__ void async_ld_b128(unsigned lds_addr, const void* gaddr) {
  asm volatile("global_load_async_to_lds_b128 %0, %1, off"
               :: "v"(lds_addr), "v"(gaddr) : "memory");
}
__device__ __forceinline__ void wait_async0() {
  asm volatile("s_wait_asynccnt 0" ::: "memory");
}

// ---------------------------------------------------------------------------
// Weight prep: conv_w (OIHW 256x256x3x3) -> bf16 [k = tap*256+ic][oc]
// ---------------------------------------------------------------------------
__global__ __launch_bounds__(256)
void wprep_conv_kernel(const float* __restrict__ w, unsigned short* __restrict__ dst) {
  int idx = blockIdx.x * 256 + threadIdx.x;        // 2304*256 total
  int k = idx >> 8, oc = idx & 255;
  int tap = k >> 8, ic = k & 255;
  dst[idx] = f2bf(w[(oc * 256 + ic) * 9 + tap]);
}

// Linear weight W[dout][din] (row major) -> bf16 B[k=din][n=dout]
__global__ __launch_bounds__(256)
void wprep_t_kernel(const float* __restrict__ src, unsigned short* __restrict__ dst,
                    int K, int N) {
  long idx = (long)blockIdx.x * 256 + threadIdx.x;
  if (idx >= (long)K * N) return;
  int k = (int)(idx / N), n = (int)(idx % N);
  dst[idx] = f2bf(src[(long)n * K + k]);
}

// ---------------------------------------------------------------------------
// Fused ROI bilinear sample + 3x3 SAME conv (implicit GEMM, WMMA bf16)
// + conv bias + BN + ReLU + spatial mean.  One workgroup (8 waves) per ROI.
// ---------------------------------------------------------------------------
__global__ __launch_bounds__(256)
void roi_conv_kernel(const float* __restrict__ world, const float* __restrict__ rois,
                     const unsigned short* __restrict__ wconv,
                     const float* __restrict__ convb, const float* __restrict__ bng,
                     const float* __restrict__ bnb, const float* __restrict__ bnm,
                     const float* __restrict__ bnv,
                     float* __restrict__ xf, unsigned short* __restrict__ xb) {
  __shared__ __align__(16) unsigned short feats[256 * 81]; // [c][9][9] padded, bf16
  __shared__ __align__(16) unsigned short btile[32 * 256]; // one k-tile of weights
  __shared__ int   ipx[49], ipy[49];
  __shared__ float fwx[49], fwy[49];

  const int ri  = blockIdx.x;          // roi index 0..599
  const int bb  = ri / 300;            // batch
  const int tid = threadIdx.x;

  // ---- per-point bilinear geometry (matches reference exactly) ----
  if (tid < 49) {
    int py = tid / 7, px = tid % 7;
    float gx = (px - 3) / 3.0f, gy = (py - 3) / 3.0f;   // linspace(-1,1,7)
    float x1 = rois[ri * 4 + 0], y1 = rois[ri * 4 + 1];
    float x2 = rois[ri * 4 + 2], y2 = rois[ri * 4 + 3];
    const float W = 200.f, H = 200.f;
    float x1n = 2.f * x1 / W - 1.f, x2n = 2.f * x2 / W - 1.f;
    float y1n = 2.f * y1 / H - 1.f, y2n = 2.f * y2 / H - 1.f;
    float sgx = x1n + (gx + 1.f) * 0.5f * (x2n - x1n);
    float sgy = y1n + (gy + 1.f) * 0.5f * (y2n - y1n);
    float ix = (sgx + 1.f) * 0.5f * (W - 1.f);
    float iy = (sgy + 1.f) * 0.5f * (H - 1.f);
    float x0 = floorf(ix), y0 = floorf(iy);
    ipx[tid] = (int)x0; ipy[tid] = (int)y0;
    fwx[tid] = ix - x0; fwy[tid] = iy - y0;
  }
  for (int i = tid; i < 256 * 81; i += 256) feats[i] = 0;  // zero incl. SAME pad
  __syncthreads();

  // ---- gather (L2-served; world fits in 192MB L2) -> bf16 LDS tile ----
  for (int i = tid; i < 49 * 256; i += 256) {
    int p = i >> 8, c = i & 255;
    int x0 = ipx[p], y0 = ipy[p];
    float wx = fwx[p], wy = fwy[p];
    const float* base = world + ((size_t)bb * 256 + c) * 40000;
    auto g = [&](int y, int x) -> float {
      if (x < 0 || x > 199 || y < 0 || y > 199) return 0.f;
      return base[y * 200 + x];
    };
    float v = g(y0, x0)     * (1.f - wy) * (1.f - wx)
            + g(y0, x0 + 1) * (1.f - wy) * wx
            + g(y0 + 1, x0) * wy * (1.f - wx)
            + g(y0 + 1, x0 + 1) * wy * wx;
    int py = p / 7, px = p % 7;
    feats[c * 81 + (py + 1) * 9 + (px + 1)] = f2bf(v);
  }

  // ---- implicit-GEMM conv: [64(pos,pad) x 2304] x [2304 x 256] ----
  const int w = tid >> 5, lane = tid & 31;
  const int r = lane & 15, h = lane >> 4;
  const unsigned btile_lds = lds_off(btile);
  v8f acc[4][2] = {};

  for (int kt = 0; kt < 72; ++kt) {                 // 72 k-tiles of 32
    __syncthreads();
    {
      // async 16KB weight k-tile: 1024 x B128, 4 per thread
      const unsigned short* gsrc = wconv + kt * 8192;
#pragma unroll
      for (int it = 0; it < 4; ++it) {
        int i = tid + it * 256;
        async_ld_b128(btile_lds + i * 16, gsrc + i * 8);
      }
      wait_async0();
    }
    __syncthreads();
    const int tap = kt >> 3;
    const int kh = tap / 3, kw = tap % 3;
    const int cbase = (kt & 7) * 32;

    Frag16 b0, b1;
#pragma unroll
    for (int e = 0; e < 16; ++e) {                  // B frag: elem e = B[16h+e][n]
      b0.u[e] = btile[(h * 16 + e) * 256 + (w * 32 + r)];
      b1.u[e] = btile[(h * 16 + e) * 256 + (w * 32 + 16 + r)];
    }
#pragma unroll
    for (int mt = 0; mt < 4; ++mt) {
      int m = mt * 16 + r;
      Frag16 a;
      if (m < 49) {
        int py = m / 7, px = m % 7;
        const unsigned short* fp = &feats[(py + kh) * 9 + (px + kw)];
#pragma unroll
        for (int v2 = 0; v2 < 8; ++v2) {            // A frag: elem e = A[m][16h+e]
          int k0 = 16 * h + 2 * v2;
          int c = cbase + k0;
          a.u[2 * v2]     = fp[c * 81];
          a.u[2 * v2 + 1] = fp[(c + 1) * 81];
        }
      } else {
#pragma unroll
        for (int e = 0; e < 16; ++e) a.u[e] = 0;
      }
      acc[mt][0] = __builtin_amdgcn_wmma_f32_16x16x32_bf16(
          false, a.v, false, b0.v, (short)0, acc[mt][0], false, false);
      acc[mt][1] = __builtin_amdgcn_wmma_f32_16x16x32_bf16(
          false, a.v, false, b1.v, (short)0, acc[mt][1], false, false);
    }
  }

  // ---- epilogue: conv bias + BN + ReLU + mean over 49 positions ----
#pragma unroll
  for (int j = 0; j < 2; ++j) {
    int oc = w * 32 + j * 16 + r;
    float inv = rsqrtf(bnv[oc] + 1e-5f);
    float gsc = bng[oc] * inv;
    float bia = convb[oc], sub = bnm[oc], add = bnb[oc];
    float s = 0.f;
#pragma unroll
    for (int mt = 0; mt < 4; ++mt) {
#pragma unroll
      for (int q = 0; q < 8; ++q) {
        int m = mt * 16 + q + 8 * h;                // D layout: row = r'+8h
        if (m < 49) {
          float val = (acc[mt][j][q] + bia - sub) * gsc + add;
          s += fmaxf(val, 0.f);
        }
      }
    }
    s += __shfl_xor(s, 16, 32);                      // combine wave halves
    if (h == 0) {
      float mean = s * (1.0f / 49.0f);
      xf[ri * 256 + oc] = mean;
      xb[ri * 256 + oc] = f2bf(mean);
    }
  }
}

// ---------------------------------------------------------------------------
// Generic batched bf16 WMMA GEMM: C = scale*(A@B) + bias, optional ReLU.
// A: [M x K] bf16 row-major (lda), B: [K x N] bf16 row-major (ldb).
// 64x64 block tile, 8 waves, each wave = 1 M-tile x 2 N-tiles.
// B tile staged via async-to-LDS B128 when fully in-bounds & 16B-aligned.
// ---------------------------------------------------------------------------
__global__ __launch_bounds__(256)
void gemm_bf16_kernel(const unsigned short* __restrict__ A, long sA, int lda,
                      const unsigned short* __restrict__ B, long sB, int ldb,
                      const float* __restrict__ bias,
                      float* __restrict__ Cf, unsigned short* __restrict__ Cb,
                      long sC, int ldc, int M, int N, int K,
                      float scale, int relu) {
  __shared__ __align__(16) unsigned short As[64][34];   // padded (conflict-free A frags)
  __shared__ __align__(16) unsigned short Bs[32 * 64];  // row stride 128B (async B128)

  const int batch = blockIdx.z;
  A += (long)batch * sA;
  B += (long)batch * sB;
  const int mBase = blockIdx.y * 64, nBase = blockIdx.x * 64;
  const int tid = threadIdx.x, w = tid >> 5, lane = tid & 31;
  const int mt = w & 3, ntb = (w >> 2) * 2;
  const int r = lane & 15, h = lane >> 4;
  const bool b_aligned = ((ldb & 7) == 0) && (nBase + 64 <= N);
  const unsigned Bs_lds = lds_off(Bs);

  v8f acc0 = {}, acc1 = {};

  for (int kk = 0; kk < K; kk += 32) {
    __syncthreads();
    for (int i = tid; i < 64 * 32; i += 256) {
      int m = i >> 5, k = i & 31;
      int gm = mBase + m, gk = kk + k;
      As[m][k] = (gm < M && gk < K) ? A[(long)gm * lda + gk] : (unsigned short)0;
    }
    bool use_async = b_aligned && (kk + 32 <= K);
    if (use_async) {
      // 4KB B tile: 256 x B128 (one per thread), ASYNCcnt path (no VGPR bounce)
      int krow = tid >> 3, seg = tid & 7;
      async_ld_b128(Bs_lds + (unsigned)tid * 16,
                    B + (long)(kk + krow) * ldb + nBase + seg * 8);
    } else {
      for (int i = tid; i < 32 * 64; i += 256) {
        int k = i >> 6, n = i & 63;
        int gk = kk + k, gn = nBase + n;
        Bs[k * 64 + n] = (gk < K && gn < N) ? B[(long)gk * ldb + gn]
                                            : (unsigned short)0;
      }
    }
    if (kk + 32 < K) {
      __builtin_prefetch((const void*)&B[(long)(kk + 32) * ldb + nBase], 0, 0);
      __builtin_prefetch((const void*)&A[(long)mBase * lda + kk + 32], 0, 0);
    }
    if (use_async) wait_async0();                    // drain ASYNCcnt before barrier
    __syncthreads();

    Frag16 a;
#pragma unroll
    for (int v2 = 0; v2 < 8; ++v2) {
      int k0 = 16 * h + 2 * v2;
      a.u[2 * v2]     = As[mt * 16 + r][k0];
      a.u[2 * v2 + 1] = As[mt * 16 + r][k0 + 1];
    }
    Frag16 b0, b1;
#pragma unroll
    for (int e = 0; e < 16; ++e) {
      b0.u[e] = Bs[(h * 16 + e) * 64 + ntb * 16 + r];
      b1.u[e] = Bs[(h * 16 + e) * 64 + (ntb + 1) * 16 + r];
    }
    acc0 = __builtin_amdgcn_wmma_f32_16x16x32_bf16(false, a.v, false, b0.v,
                                                   (short)0, acc0, false, false);
    acc1 = __builtin_amdgcn_wmma_f32_16x16x32_bf16(false, a.v, false, b1.v,
                                                   (short)0, acc1, false, false);
  }

#pragma unroll
  for (int j = 0; j < 2; ++j) {
    v8f acc = j ? acc1 : acc0;
    int gn = nBase + (ntb + j) * 16 + r;
    if (gn >= N) continue;
    float bv = bias ? bias[gn] : 0.f;
#pragma unroll
    for (int q = 0; q < 8; ++q) {
      int gm = mBase + mt * 16 + q + 8 * h;
      if (gm >= M) continue;
      float val = acc[q] * scale + bv;
      if (relu) val = fmaxf(val, 0.f);
      long off = (long)batch * sC + (long)gm * ldc + gn;
      if (Cf) Cf[off] = val;
      if (Cb) Cb[off] = f2bf(val);
    }
  }
}

// ---------------------------------------------------------------------------
// qkv [600x768] bf16 -> q[bh][t][d], kT[bh][d][t], v[bh][t][d]  (bh = b*8+h)
// ---------------------------------------------------------------------------
__global__ __launch_bounds__(256)
void qkv_split_kernel(const unsigned short* __restrict__ qkv,
                      unsigned short* __restrict__ q, unsigned short* __restrict__ kT,
                      unsigned short* __restrict__ v) {
  int idx = blockIdx.x * 256 + threadIdx.x;
  if (idx >= 600 * 768) return;
  int tk = idx / 768, col = idx % 768;
  int b = tk / 300, t = tk % 300;
  unsigned short val = qkv[idx];
  if (col < 256) {
    int hd = col >> 5, d = col & 31;
    q[(((b * 8 + hd) * 300) + t) * 32 + d] = val;
  } else if (col < 512) {
    int c = col - 256, hd = c >> 5, d = c & 31;
    kT[((b * 8 + hd) * 32 + d) * 300 + t] = val;   // transposed for [K][N] GEMM
  } else {
    int c = col - 512, hd = c >> 5, d = c & 31;
    v[(((b * 8 + hd) * 300) + t) * 32 + d] = val;
  }
}

// softmax over rows of length `cols` (scores already scaled by 1/sqrt(dh))
__global__ __launch_bounds__(128)
void softmax_kernel(const float* __restrict__ S, unsigned short* __restrict__ A,
                    int cols) {
  __shared__ float red[128];
  long row = blockIdx.x;
  const float* s = S + row * cols;
  unsigned short* a = A + row * cols;
  int t = threadIdx.x;
  float m = -3.0e38f;
  for (int i = t; i < cols; i += 128) m = fmaxf(m, s[i]);
  red[t] = m; __syncthreads();
  for (int o = 64; o > 0; o >>= 1) { if (t < o) red[t] = fmaxf(red[t], red[t + o]); __syncthreads(); }
  m = red[0]; __syncthreads();
  float sum = 0.f;
  for (int i = t; i < cols; i += 128) sum += __expf(s[i] - m);
  red[t] = sum; __syncthreads();
  for (int o = 64; o > 0; o >>= 1) { if (t < o) red[t] += red[t + o]; __syncthreads(); }
  float inv = 1.f / red[0];
  for (int i = t; i < cols; i += 128) a[i] = f2bf(__expf(s[i] - m) * inv);
}

// attn_out [bh][t][d] f32 -> o [600x256] bf16
__global__ __launch_bounds__(256)
void orearr_kernel(const float* __restrict__ attno, unsigned short* __restrict__ o) {
  int idx = blockIdx.x * 256 + threadIdx.x;  // < 153600
  int tk = idx >> 8, col = idx & 255;
  int b = tk / 300, t = tk % 300, hd = col >> 5, d = col & 31;
  o[idx] = f2bf(attno[(((b * 8 + hd) * 300) + t) * 32 + d]);
}

// out = LayerNorm(x + delta)*g + b ; writes f32 and bf16 copies. 1 block / row.
__global__ __launch_bounds__(256)
void addln_kernel(const float* __restrict__ x, const float* __restrict__ d,
                  const float* __restrict__ g, const float* __restrict__ b,
                  float* __restrict__ outf, unsigned short* __restrict__ outb) {
  __shared__ float red[256];
  int row = blockIdx.x, t = threadIdx.x;
  float v = x[row * 256 + t] + d[row * 256 + t];
  red[t] = v; __syncthreads();
  for (int o = 128; o > 0; o >>= 1) { if (t < o) red[t] += red[t + o]; __syncthreads(); }
  float mu = red[0] * (1.f / 256.f);
  __syncthreads();
  float c = v - mu;
  red[t] = c * c; __syncthreads();
  for (int o = 128; o > 0; o >>= 1) { if (t < o) red[t] += red[t + o]; __syncthreads(); }
  float var = red[0] * (1.f / 256.f);
  float y = c * rsqrtf(var + 1e-5f) * g[t] + b[t];
  outf[row * 256 + t] = y;
  outb[row * 256 + t] = f2bf(y);
}

// ---------------------------------------------------------------------------
static inline int cdiv(int a, int b) { return (a + b - 1) / b; }

extern "C" void kernel_launch(void* const* d_in, const int* in_sizes, int n_in,
                              void* d_out, int out_size, void* d_ws, size_t ws_size,
                              hipStream_t stream) {
  (void)in_sizes; (void)n_in; (void)out_size; (void)ws_size;

  const float* world = (const float*)d_in[0];
  const float* rois  = (const float*)d_in[1];
  const float* convw = (const float*)d_in[2];
  const float* convb = (const float*)d_in[3];
  const float* bng   = (const float*)d_in[4];
  const float* bnb   = (const float*)d_in[5];
  const float* bnm   = (const float*)d_in[6];
  const float* bnv   = (const float*)d_in[7];

  char* wsp = (char*)d_ws;
  auto alloc = [&](size_t bytes) -> void* {
    void* p = (void*)wsp;
    wsp += (bytes + 255) & ~(size_t)255;
    return p;
  };

  unsigned short* wconv  = (unsigned short*)alloc((size_t)2304 * 256 * 2);
  float*          xf     = (float*)alloc((size_t)600 * 256 * 4);
  unsigned short* xb     = (unsigned short*)alloc((size_t)600 * 256 * 2);
  unsigned short* qkv    = (unsigned short*)alloc((size_t)600 * 768 * 2);
  unsigned short* qb     = (unsigned short*)alloc((size_t)153600 * 2);
  unsigned short* kT     = (unsigned short*)alloc((size_t)153600 * 2);
  unsigned short* vb     = (unsigned short*)alloc((size_t)153600 * 2);
  float*          scores = (float*)alloc((size_t)16 * 300 * 300 * 4);
  unsigned short* attn   = (unsigned short*)alloc((size_t)16 * 300 * 300 * 2);
  float*          attno  = (float*)alloc((size_t)153600 * 4);
  unsigned short* ob     = (unsigned short*)alloc((size_t)153600 * 2);
  float*          tmpf   = (float*)alloc((size_t)600 * 256 * 4);
  unsigned short* hid    = (unsigned short*)alloc((size_t)600 * 512 * 2);
  unsigned short* wT_in  = (unsigned short*)alloc((size_t)256 * 768 * 2);
  unsigned short* wT_out = (unsigned short*)alloc((size_t)256 * 256 * 2);
  unsigned short* wT_f1  = (unsigned short*)alloc((size_t)256 * 512 * 2);
  unsigned short* wT_f2  = (unsigned short*)alloc((size_t)512 * 256 * 2);
  unsigned short* wT_h1  = (unsigned short*)alloc((size_t)256 * 512 * 2);
  unsigned short* wT_h2  = (unsigned short*)alloc((size_t)512 * 98304 * 2);

  float* out_x    = (float*)d_out;                       // (2,300,256)
  float* out_pose = out_x + (size_t)600 * 256;           // (2,300,18,3)
  float* out_mesh = out_pose + (size_t)600 * 54;         // (2,300,3,32,32,32)
  float* out_traj = out_mesh + (size_t)600 * 98304;      // (2,300,6,12,3)

  // ---- conv weight prep + fused ROI-sample/conv/BN/ReLU/mean ----
  wprep_conv_kernel<<<2304, 256, 0, stream>>>(convw, wconv);
  roi_conv_kernel<<<600, 256, 0, stream>>>(world, rois, wconv, convb,
                                           bng, bnb, bnm, bnv, xf, xb);

  const float iscale = 0.17677669529663687f;  // 1/sqrt(32)

  for (int l = 0; l < 4; ++l) {
    const float* in_w  = (const float*)d_in[8 + 12 * l + 0];
    const float* in_b  = (const float*)d_in[8 + 12 * l + 1];
    const float* out_w = (const float*)d_in[8 + 12 * l + 2];
    const float* out_b = (const float*)d_in[8 + 12 * l + 3];
    const float* ln1g  = (const float*)d_in[8 + 12 * l + 4];
    const float* ln1b  = (const float*)d_in[8 + 12 * l + 5];
    const float* ffw1  = (const float*)d_in[8 + 12 * l + 6];
    const float* ffb1  = (const float*)d_in[8 + 12 * l + 7];
    const float* ffw2  = (const float*)d_in[8 + 12 * l + 8];
    const float* ffb2  = (const float*)d_in[8 + 12 * l + 9];
    const float* ln2g  = (const float*)d_in[8 + 12 * l + 10];
    const float* ln2b  = (const float*)d_in[8 + 12 * l + 11];

    wprep_t_kernel<<<cdiv(256 * 768, 256), 256, 0, stream>>>(in_w, wT_in, 256, 768);
    wprep_t_kernel<<<cdiv(256 * 256, 256), 256, 0, stream>>>(out_w, wT_out, 256, 256);
    wprep_t_kernel<<<cdiv(256 * 512, 256), 256, 0, stream>>>(ffw1, wT_f1, 256, 512);
    wprep_t_kernel<<<cdiv(512 * 256, 256), 256, 0, stream>>>(ffw2, wT_f2, 512, 256);

    // qkv = x @ in_w.T + in_b  -> bf16
    gemm_bf16_kernel<<<dim3(cdiv(768, 64), cdiv(600, 64), 1), 256, 0, stream>>>(
        xb, 0, 256, wT_in, 0, 768, in_b, nullptr, qkv, 0, 768,
        600, 768, 256, 1.f, 0);
    qkv_split_kernel<<<1800, 256, 0, stream>>>(qkv, qb, kT, vb);

    // scores[bh] = q @ k^T / sqrt(dh)   (batch = 16)
    gemm_bf16_kernel<<<dim3(5, 5, 16), 256, 0, stream>>>(
        qb, 9600, 32, kT, 9600, 300, nullptr, scores, nullptr, 90000, 300,
        300, 300, 32, iscale, 0);
    softmax_kernel<<<4800, 128, 0, stream>>>(scores, attn, 300);

    // out[bh] = attn @ v
    gemm_bf16_kernel<<<dim3(1, 5, 16), 256, 0, stream>>>(
        attn, 90000, 300, vb, 9600, 32, nullptr, attno, nullptr, 9600, 32,
        300, 32, 300, 1.f, 0);
    orearr_kernel<<<600, 256, 0, stream>>>(attno, ob);

    // out projection + residual LN1
    gemm_bf16_kernel<<<dim3(4, 10, 1), 256, 0, stream>>>(
        ob, 0, 256, wT_out, 0, 256, out_b, tmpf, nullptr, 0, 256,
        600, 256, 256, 1.f, 0);
    addln_kernel<<<600, 256, 0, stream>>>(xf, tmpf, ln1g, ln1b, xf, xb);

    // FFN
    gemm_bf16_kernel<<<dim3(8, 10, 1), 256, 0, stream>>>(
        xb, 0, 256, wT_f1, 0, 512, ffb1, nullptr, hid, 0, 512,
        600, 512, 256, 1.f, 1);
    gemm_bf16_kernel<<<dim3(4, 10, 1), 256, 0, stream>>>(
        hid, 0, 512, wT_f2, 0, 256, ffb2, tmpf, nullptr, 0, 256,
        600, 256, 512, 1.f, 0);
    addln_kernel<<<600, 256, 0, stream>>>(xf, tmpf, ln2g, ln2b,
                                          (l == 3) ? out_x : xf, xb);
  }

  // ---- heads ----
  // pose: 256 -> 256 relu -> 54
  wprep_t_kernel<<<cdiv(256 * 256, 256), 256, 0, stream>>>(
      (const float*)d_in[56], wT_h1, 256, 256);
  gemm_bf16_kernel<<<dim3(4, 10, 1), 256, 0, stream>>>(
      xb, 0, 256, wT_h1, 0, 256, (const float*)d_in[57], nullptr, hid, 0, 256,
      600, 256, 256, 1.f, 1);
  wprep_t_kernel<<<cdiv(256 * 54, 256), 256, 0, stream>>>(
      (const float*)d_in[58], wT_h2, 256, 54);
  gemm_bf16_kernel<<<dim3(1, 10, 1), 256, 0, stream>>>(
      hid, 0, 256, wT_h2, 0, 54, (const float*)d_in[59], out_pose, nullptr, 0, 54,
      600, 54, 256, 1.f, 0);

  // mesh: 256 -> 512 relu -> 98304  (dominant, HBM-bound GEMM -> async path)
  wprep_t_kernel<<<cdiv(256 * 512, 256), 256, 0, stream>>>(
      (const float*)d_in[60], wT_h1, 256, 512);
  gemm_bf16_kernel<<<dim3(8, 10, 1), 256, 0, stream>>>(
      xb, 0, 256, wT_h1, 0, 512, (const float*)d_in[61], nullptr, hid, 0, 512,
      600, 512, 256, 1.f, 1);
  wprep_t_kernel<<<cdiv(512 * 98304, 256), 256, 0, stream>>>(
      (const float*)d_in[62], wT_h2, 512, 98304);
  gemm_bf16_kernel<<<dim3(1536, 10, 1), 256, 0, stream>>>(
      hid, 0, 512, wT_h2, 0, 98304, (const float*)d_in[63], out_mesh, nullptr, 0, 98304,
      600, 98304, 512, 1.f, 0);

  // traj: 256 -> 512 relu -> 216
  wprep_t_kernel<<<cdiv(256 * 512, 256), 256, 0, stream>>>(
      (const float*)d_in[64], wT_h1, 256, 512);
  gemm_bf16_kernel<<<dim3(8, 10, 1), 256, 0, stream>>>(
      xb, 0, 256, wT_h1, 0, 512, (const float*)d_in[65], nullptr, hid, 0, 512,
      600, 512, 256, 1.f, 1);
  wprep_t_kernel<<<cdiv(512 * 216, 256), 256, 0, stream>>>(
      (const float*)d_in[66], wT_h2, 512, 216);
  gemm_bf16_kernel<<<dim3(4, 10, 1), 256, 0, stream>>>(
      hid, 0, 512, wT_h2, 0, 216, (const float*)d_in[67], out_traj, nullptr, 0, 216,
      600, 216, 512, 1.f, 0);
}
